// Mutihead_Attention_K_36833639530819
// MI455X (gfx1250) — matrix operand order repaired and checked by reading, not compiled, not run on hardware
//
#include <hip/hip_runtime.h>
#include <hip/hip_bf16.h>
#include <stddef.h>
#include <stdint.h>

// ---------------------------------------------------------------------------
// Multi-head attention variant for MI455X (gfx1250, wave32, WMMA).
// All GEMMs use v_wmma_f32_16x16x32_f16 (f16 inputs, f32 accumulate).
// Projection GEMMs stage the weight tile in LDS via async-to-LDS copies
// (ASYNCcnt / s_wait_asynccnt, double buffered). Softmax statistics in f32.
// ---------------------------------------------------------------------------

typedef __attribute__((ext_vector_type(8)))  float    v8f;
typedef __attribute__((ext_vector_type(16))) _Float16 v16h;
typedef __attribute__((ext_vector_type(8)))  _Float16 v8h;
typedef __attribute__((ext_vector_type(4)))  _Float16 v4h;

#define H_HEADS 16
#define BATCH   4
#define SEQ     1024
#define DMODEL  1024
#define DHEAD   64
#define HB      (H_HEADS * BATCH)          // 64 slices
#define NE_ACT  (BATCH * SEQ * DMODEL)     // 4,194,304 activations
#define NE_W    (DMODEL * DMODEL)          // 1,048,576 weights
#define SLICE_QK 65536                     // SEQ * DHEAD
#define SLICE_S  1048576                   // SEQ * SEQ
#define NORM_FACT 0.03125f                 // 1/sqrt(1024)

// ---------------- WMMA fragment helpers (CDNA5 16x16x32 f16) ----------------

__device__ __forceinline__ v8f wmma_f16(v16h a, v16h b, v8f c) {
  // 8 args: (neg_a, A, neg_b, B, c_mod, C, reuse_a, reuse_b)
  return __builtin_amdgcn_wmma_f32_16x16x32_f16(false, a, false, b, (short)0, c,
                                                false, false);
}

// A tile 16x32 (MxK) at tile origin `A`, row-major with leading dim lda (f16).
// Lane l<16: row l, K = {0..7, 16..23}; lane>=16: row l-16, K = {8..15, 24..31}.
__device__ __forceinline__ v16h load_a_frag(const _Float16* __restrict__ A,
                                            int lda, int lane) {
  int h = lane >> 4, l = lane & 15;
  const _Float16* p = A + (size_t)l * lda + h * 8;
  v8h lo = *(const v8h*)p;
  v8h hi = *(const v8h*)(p + 16);
  v16h r;
#pragma unroll
  for (int i = 0; i < 8; ++i) { r[i] = lo[i]; r[i + 8] = hi[i]; }
  return r;
}

// B tile 32x16 (KxN); storage is N rows x K cols row-major (K contiguous).
// Bp points at element (k0, n0) i.e. Bstore + n0*ldb + k0.
// Lane l<16: col l, K = {0..7, 16..23}; lane>=16: col l-16, K = {8..15, 24..31}.
__device__ __forceinline__ v16h load_b_frag(const _Float16* __restrict__ Bp,
                                            int ldb, int lane) {
  int h = lane >> 4, n = lane & 15;
  const _Float16* p = Bp + (size_t)n * ldb + h * 8;
  v8h lo = *(const v8h*)p;
  v8h hi = *(const v8h*)(p + 16);
  v16h r;
#pragma unroll
  for (int i = 0; i < 8; ++i) { r[i] = lo[i]; r[i + 8] = hi[i]; }
  return r;
}

// Async global -> LDS copy, 16 bytes per lane. lds_off is the raw LDS byte
// offset (low 32 bits of a flat LDS address, per ISA 10.2 aperture rules).
__device__ __forceinline__ void lds_async_copy_b128(unsigned lds_off,
                                                    unsigned long long gaddr) {
  asm volatile("global_load_async_to_lds_b128 %0, %1, off"
               :: "v"(lds_off), "v"(gaddr)
               : "memory");
}

__device__ __forceinline__ float wave_sum(float v) {
#pragma unroll
  for (int o = 16; o; o >>= 1) v += __shfl_xor(v, o, 32);
  return v;
}
__device__ __forceinline__ float wave_max(float v) {
#pragma unroll
  for (int o = 16; o; o >>= 1) v = fmaxf(v, __shfl_xor(v, o, 32));
  return v;
}

// ---------------- f32 -> f16 conversion (4 elems / thread) ------------------

__global__ void cvt_f16_kernel(const float* __restrict__ in,
                               _Float16* __restrict__ out, int n4) {
  int i = blockIdx.x * blockDim.x + threadIdx.x;
  if (i < n4) {
    float4 v = ((const float4*)in)[i];
    v4h o;
    o[0] = (_Float16)v.x; o[1] = (_Float16)v.y;
    o[2] = (_Float16)v.z; o[3] = (_Float16)v.w;
    ((v4h*)out)[i] = o;
  }
}

// ---------------- Generic GEMM: C = A(MxK) * W(NxK)^T + bias ----------------
// Weight tile (64 x 32 f16 = 4 KB) is double-buffered in LDS via async copies.
// MODE 0: store f16 row-major [M,N]
// MODE 1: store f16 scattered for V: flat=(r*N+c) -> [hb][d][t]  (N must be 1024)
// MODE 2: store f32 row-major [M,N]  (final output projection)
// Block = 128 threads (4 waves), block tile 64x64, wave tile 32x32.

template <int MODE>
__global__ void gemm_bias_kernel(const _Float16* __restrict__ A,
                                 const _Float16* __restrict__ W,
                                 const float* __restrict__ bias,
                                 void* __restrict__ Cout,
                                 int M, int N, int K) {
  __shared__ _Float16 Bsh[2][64 * 32];   // 2 x 4 KB weight tiles

  int tid = threadIdx.x;
  int lane = tid & 31, wid = tid >> 5;
  int wy = wid >> 1, wx = wid & 1;
  int rbase = blockIdx.y * 64 + wy * 32;
  int cblock = blockIdx.x * 64;
  int cbase = cblock + wx * 32;

  // 2 x 16B async chunks per thread cover the 64x32 weight tile.
  int chunk = tid * 2;            // 0,2,...,254
  int brow  = chunk >> 2;         // 0..63 (weight row within tile)
  int boff  = (chunk & 3) * 8;    // 0 or 16 halves within the 32-half row

  auto issue_tile = [&](int kk, int buf) {
    const _Float16* g = W + (size_t)(cblock + brow) * K + kk + boff;
    unsigned d = (unsigned)(uintptr_t)&Bsh[buf][brow * 32 + boff];
    lds_async_copy_b128(d, (unsigned long long)(uintptr_t)g);
    lds_async_copy_b128(d + 16, (unsigned long long)(uintptr_t)(g + 8));
  };

  v8f zero = {0.f, 0.f, 0.f, 0.f, 0.f, 0.f, 0.f, 0.f};
  v8f acc[2][2];
#pragma unroll
  for (int i = 0; i < 2; ++i)
#pragma unroll
    for (int j = 0; j < 2; ++j) acc[i][j] = zero;

  issue_tile(0, 0);                        // prologue copy

  for (int kk = 0; kk < K; kk += 32) {
    int buf = (kk >> 5) & 1;
    bool more = (kk + 32) < K;
    if (more) {
      issue_tile(kk + 32, buf ^ 1);        // overlap next weight tile
      asm volatile("s_wait_asynccnt 0x2" ::: "memory");  // tile kk done
    } else {
      asm volatile("s_wait_asynccnt 0x0" ::: "memory");
    }
    __syncthreads();                       // LDS writes visible to all waves

    const _Float16* Ab = A + (size_t)rbase * K + kk;
    if (more) __builtin_prefetch(Ab + 32, 0, 1);   // global_prefetch_b8
    v16h a0 = load_a_frag(Ab, K, lane);
    v16h a1 = load_a_frag(Ab + (size_t)16 * K, K, lane);
    v16h b0 = load_b_frag(&Bsh[buf][(wx * 32 +  0) * 32], 32, lane);
    v16h b1 = load_b_frag(&Bsh[buf][(wx * 32 + 16) * 32], 32, lane);
    acc[0][0] = wmma_f16(a0, b0, acc[0][0]);
    acc[0][1] = wmma_f16(a0, b1, acc[0][1]);
    acc[1][0] = wmma_f16(a1, b0, acc[1][0]);
    acc[1][1] = wmma_f16(a1, b1, acc[1][1]);

    __syncthreads();                       // protect buffer before next issue
  }

  int h = lane >> 4, n = lane & 15;
#pragma unroll
  for (int i = 0; i < 2; ++i)
#pragma unroll
    for (int j = 0; j < 2; ++j) {
      int c = cbase + j * 16 + n;
      float bv = bias[c];
#pragma unroll
      for (int k = 0; k < 8; ++k) {
        int r = rbase + i * 16 + h * 8 + k;
        float val = acc[i][j][k] + bv;
        if (MODE == 0) {
          ((_Float16*)Cout)[(size_t)r * N + c] = (_Float16)val;
        } else if (MODE == 2) {
          ((float*)Cout)[(size_t)r * N + c] = val;
        } else {  // MODE == 1 : V-transpose scatter, N == 1024
          size_t flat = (size_t)r * 1024 + c;
          size_t hb = flat >> 16;
          size_t t  = (flat >> 6) & 1023;
          size_t d  = flat & 63;
          ((_Float16*)Cout)[hb * (size_t)SLICE_QK + d * 1024 + t] = (_Float16)val;
        }
      }
    }
}

// ---------------- Score: S[hb][s][t] = (Q_hb · K_hb^T) / 32 -----------------

__global__ void score_kernel(const _Float16* __restrict__ Qh,
                             const _Float16* __restrict__ Kh,
                             float* __restrict__ S) {
  int hb = blockIdx.z;
  int lane = threadIdx.x & 31, wid = threadIdx.x >> 5;
  int wy = wid >> 1, wx = wid & 1;
  int rbase = blockIdx.y * 64 + wy * 32;   // s
  int cbase = blockIdx.x * 64 + wx * 32;   // t
  const _Float16* Qs = Qh + (size_t)hb * SLICE_QK;
  const _Float16* Ks = Kh + (size_t)hb * SLICE_QK;

  v8f zero = {0.f, 0.f, 0.f, 0.f, 0.f, 0.f, 0.f, 0.f};
  v8f acc[2][2];
#pragma unroll
  for (int i = 0; i < 2; ++i)
#pragma unroll
    for (int j = 0; j < 2; ++j) acc[i][j] = zero;

#pragma unroll
  for (int kk = 0; kk < DHEAD; kk += 32) {
    v16h a0 = load_a_frag(Qs + (size_t)(rbase +  0) * DHEAD + kk, DHEAD, lane);
    v16h a1 = load_a_frag(Qs + (size_t)(rbase + 16) * DHEAD + kk, DHEAD, lane);
    v16h b0 = load_b_frag(Ks + (size_t)(cbase +  0) * DHEAD + kk, DHEAD, lane);
    v16h b1 = load_b_frag(Ks + (size_t)(cbase + 16) * DHEAD + kk, DHEAD, lane);
    acc[0][0] = wmma_f16(a0, b0, acc[0][0]);
    acc[0][1] = wmma_f16(a0, b1, acc[0][1]);
    acc[1][0] = wmma_f16(a1, b0, acc[1][0]);
    acc[1][1] = wmma_f16(a1, b1, acc[1][1]);
  }

  float* Ss = S + (size_t)hb * SLICE_S;
  int h = lane >> 4, n = lane & 15;
#pragma unroll
  for (int i = 0; i < 2; ++i)
#pragma unroll
    for (int j = 0; j < 2; ++j)
#pragma unroll
      for (int k = 0; k < 8; ++k) {
        int r = rbase + i * 16 + h * 8 + k;
        int c = cbase + j * 16 + n;
        Ss[(size_t)r * SEQ + c] = acc[i][j][k] * NORM_FACT;
      }
}

// -------- Column stats (softmax over query axis): max & 1/sumexp per t ------

__global__ void colstats_kernel(const float* __restrict__ S,
                                float* __restrict__ cmax,
                                float* __restrict__ crcp) {
  int hb = blockIdx.y;
  int t = blockIdx.x * 256 + threadIdx.x;
  const float* Ss = S + (size_t)hb * SLICE_S + t;
  float m = -INFINITY, l = 0.f;
  for (int s = 0; s < SEQ; ++s) {
    float v = Ss[(size_t)s * SEQ];
    float nm = fmaxf(m, v);
    l = l * __expf(m - nm) + __expf(v - nm);
    m = nm;
  }
  cmax[hb * SEQ + t] = m;
  crcp[hb * SEQ + t] = 1.f / l;
}

// -------- Row stats: 1/L1 (branch1), thr/max/1/Z (branch2). 1 wave / row ----

__global__ void rowstats_kernel(const float* __restrict__ S,
                                const float* __restrict__ cmax,
                                const float* __restrict__ crcp,
                                const int* __restrict__ mask,
                                float* __restrict__ invl1,
                                float* __restrict__ thrv,
                                float* __restrict__ rmxv,
                                float* __restrict__ invzv) {
  int hb = blockIdx.y;
  int lane = threadIdx.x & 31;
  int s = blockIdx.x * 8 + (threadIdx.x >> 5);
  const float* row = S + (size_t)hb * SLICE_S + (size_t)s * SEQ;
  const float* cm = cmax + hb * SEQ;
  const float* cr = crcp + hb * SEQ;

  float rs = 0.f, l1 = 0.f;
  for (int t = lane; t < SEQ; t += 32) {
    float v = row[t];
    rs += v;
    l1 += __expf(v - cm[t]) * cr[t];       // branch1 attn value (col softmax)
  }
  rs = wave_sum(rs);
  l1 = wave_sum(l1);

  float th = (mask[0] != 0) ? rs * (1.f / (float)SEQ) : -INFINITY;

  float mx = -INFINITY;
  for (int t = lane; t < SEQ; t += 32) {
    float v = row[t];
    if (v > th) mx = fmaxf(mx, v);
  }
  mx = wave_max(mx);

  float z = 0.f;
  for (int t = lane; t < SEQ; t += 32) {
    float v = row[t];
    if (v > th) z += __expf(v - mx);
  }
  z = wave_sum(z);

  if (lane == 0) {
    int idx = hb * SEQ + s;
    invl1[idx] = 1.f / l1;
    thrv[idx]  = th;
    rmxv[idx]  = mx;
    invzv[idx] = 1.f / z;
  }
}

// -------- Fused P·V: build f16 A-fragments of P on the fly, WMMA with Vt ----

__device__ __forceinline__ v16h make_p_frag(const float* __restrict__ row,
                                            const float* __restrict__ cm,
                                            const float* __restrict__ cr,
                                            float il1, float th, float mx,
                                            float iz, int kb, int h) {
  v16h r;
  int base = kb + h * 8;
#pragma unroll
  for (int half = 0; half < 2; ++half) {
    int off = base + half * 16;
#pragma unroll
    for (int i = 0; i < 8; ++i) {
      float sv = row[off + i];
      float p = __expf(sv - cm[off + i]) * cr[off + i] * il1;   // branch1/L1
      if (sv > th) p += __expf(sv - mx) * iz;                   // branch2
      r[half * 8 + i] = (_Float16)(0.5f * p);
    }
  }
  return r;
}

__global__ void pv_kernel(const float* __restrict__ S,
                          const _Float16* __restrict__ Vt,
                          const float* __restrict__ cmax,
                          const float* __restrict__ crcp,
                          const float* __restrict__ invl1,
                          const float* __restrict__ thrv,
                          const float* __restrict__ rmxv,
                          const float* __restrict__ invzv,
                          _Float16* __restrict__ O16) {
  int hb = blockIdx.y;
  int lane = threadIdx.x & 31, wid = threadIdx.x >> 5;
  int wy = wid >> 1, wx = wid & 1;
  int rbase = blockIdx.x * 64 + wy * 32;   // s
  int cbase = wx * 32;                     // d (DHEAD = 64)
  int l = lane & 15, h = lane >> 4;

  const float* Ss = S + (size_t)hb * SLICE_S;
  const float* cm = cmax + hb * SEQ;
  const float* cr = crcp + hb * SEQ;
  const _Float16* Vs = Vt + (size_t)hb * SLICE_QK;   // [d][t], ld = 1024

  int s0 = rbase + l, s1 = rbase + 16 + l;
  int i0 = hb * SEQ + s0, i1 = hb * SEQ + s1;
  float il0 = invl1[i0], th0 = thrv[i0], m0 = rmxv[i0], iz0 = invzv[i0];
  float il1v = invl1[i1], th1 = thrv[i1], m1 = rmxv[i1], iz1 = invzv[i1];
  const float* row0 = Ss + (size_t)s0 * SEQ;
  const float* row1 = Ss + (size_t)s1 * SEQ;

  v8f zero = {0.f, 0.f, 0.f, 0.f, 0.f, 0.f, 0.f, 0.f};
  v8f acc[2][2];
#pragma unroll
  for (int i = 0; i < 2; ++i)
#pragma unroll
    for (int j = 0; j < 2; ++j) acc[i][j] = zero;

  for (int kb = 0; kb < SEQ; kb += 32) {
    v16h a0 = make_p_frag(row0, cm, cr, il0, th0, m0, iz0, kb, h);
    v16h a1 = make_p_frag(row1, cm, cr, il1v, th1, m1, iz1, kb, h);
    v16h b0 = load_b_frag(Vs + (size_t)(cbase +  0) * SEQ + kb, SEQ, lane);
    v16h b1 = load_b_frag(Vs + (size_t)(cbase + 16) * SEQ + kb, SEQ, lane);
    acc[0][0] = wmma_f16(a0, b0, acc[0][0]);
    acc[0][1] = wmma_f16(a0, b1, acc[0][1]);
    acc[1][0] = wmma_f16(a1, b0, acc[1][0]);
    acc[1][1] = wmma_f16(a1, b1, acc[1][1]);
  }

  _Float16* Os = O16 + (size_t)hb * SLICE_QK;        // [s][d], ld = 64
  int n = l;
#pragma unroll
  for (int i = 0; i < 2; ++i)
#pragma unroll
    for (int j = 0; j < 2; ++j)
#pragma unroll
      for (int k = 0; k < 8; ++k) {
        int r = rbase + i * 16 + h * 8 + k;
        int c = cbase + j * 16 + n;
        Os[(size_t)r * DHEAD + c] = (_Float16)acc[i][j][k];
      }
}

// ---------------------------------------------------------------------------

extern "C" void kernel_launch(void* const* d_in, const int* in_sizes, int n_in,
                              void* d_out, int out_size, void* d_ws, size_t ws_size,
                              hipStream_t stream) {
  (void)in_sizes; (void)n_in; (void)out_size; (void)ws_size;

  const float* x  = (const float*)d_in[0];
  const float* y  = (const float*)d_in[1];
  const float* Wq = (const float*)d_in[2];
  const float* bq = (const float*)d_in[3];
  const float* Wk = (const float*)d_in[4];
  const float* bk = (const float*)d_in[5];
  const float* Wv = (const float*)d_in[6];
  const float* bv = (const float*)d_in[7];
  const float* Wo = (const float*)d_in[8];
  const float* bo = (const float*)d_in[9];
  const int* mask = (const int*)d_in[10];
  float* out = (float*)d_out;

  // Workspace carve-out (requires ~330 MB).
  char* wsp = (char*)d_ws;
  auto carve = [&](size_t bytes) -> void* {
    void* p = (void*)wsp;
    wsp += (bytes + 255) & ~(size_t)255;
    return p;
  };
  _Float16* Xh   = (_Float16*)carve((size_t)NE_ACT * 2);
  _Float16* Yh   = (_Float16*)carve((size_t)NE_ACT * 2);
  _Float16* Wq16 = (_Float16*)carve((size_t)NE_W * 2);
  _Float16* Wk16 = (_Float16*)carve((size_t)NE_W * 2);
  _Float16* Wv16 = (_Float16*)carve((size_t)NE_W * 2);
  _Float16* Wo16 = (_Float16*)carve((size_t)NE_W * 2);
  _Float16* Qh   = (_Float16*)carve((size_t)NE_ACT * 2);
  _Float16* Kh   = (_Float16*)carve((size_t)NE_ACT * 2);
  _Float16* Vt   = (_Float16*)carve((size_t)NE_ACT * 2);   // [hb][d][t]
  _Float16* Oh16 = (_Float16*)carve((size_t)NE_ACT * 2);   // [hb][s][d]
  float*    S    = (float*)carve((size_t)HB * SLICE_S * 4);  // 256 MB
  float*    cmax = (float*)carve((size_t)HB * SEQ * 4);
  float*    crcp = (float*)carve((size_t)HB * SEQ * 4);
  float*    iL1  = (float*)carve((size_t)HB * SEQ * 4);
  float*    thr  = (float*)carve((size_t)HB * SEQ * 4);
  float*    rmx  = (float*)carve((size_t)HB * SEQ * 4);
  float*    iZ   = (float*)carve((size_t)HB * SEQ * 4);

  // 1) f32 -> f16 conversions
  cvt_f16_kernel<<<NE_ACT / 4 / 256, 256, 0, stream>>>(x, Xh, NE_ACT / 4);
  cvt_f16_kernel<<<NE_ACT / 4 / 256, 256, 0, stream>>>(y, Yh, NE_ACT / 4);
  cvt_f16_kernel<<<NE_W / 4 / 256, 256, 0, stream>>>(Wq, Wq16, NE_W / 4);
  cvt_f16_kernel<<<NE_W / 4 / 256, 256, 0, stream>>>(Wk, Wk16, NE_W / 4);
  cvt_f16_kernel<<<NE_W / 4 / 256, 256, 0, stream>>>(Wv, Wv16, NE_W / 4);
  cvt_f16_kernel<<<NE_W / 4 / 256, 256, 0, stream>>>(Wo, Wo16, NE_W / 4);

  dim3 gemmGrid(DMODEL / 64, (BATCH * SEQ) / 64);   // (16, 64)

  // 2) Projections (weight tiles async-staged in LDS)
  gemm_bias_kernel<0><<<gemmGrid, 128, 0, stream>>>(Xh, Wq16, bq, Qh,
                                                    BATCH * SEQ, DMODEL, DMODEL);
  gemm_bias_kernel<0><<<gemmGrid, 128, 0, stream>>>(Yh, Wk16, bk, Kh,
                                                    BATCH * SEQ, DMODEL, DMODEL);
  gemm_bias_kernel<1><<<gemmGrid, 128, 0, stream>>>(Yh, Wv16, bv, Vt,
                                                    BATCH * SEQ, DMODEL, DMODEL);

  // 3) Scores per (h,b) slice
  score_kernel<<<dim3(SEQ / 64, SEQ / 64, HB), 128, 0, stream>>>(Qh, Kh, S);

  // 4) Column softmax stats (softmax over query axis)
  colstats_kernel<<<dim3(SEQ / 256, HB), 256, 0, stream>>>(S, cmax, crcp);

  // 5) Row stats: L1 norm (branch1), mean-threshold masked softmax (branch2)
  rowstats_kernel<<<dim3(SEQ / 8, HB), 256, 0, stream>>>(S, cmax, crcp, mask,
                                                         iL1, thr, rmx, iZ);

  // 6) Fused 0.5*(attn+attn1) @ V
  pv_kernel<<<dim3(SEQ / 64, HB), 128, 0, stream>>>(S, Vt, cmax, crcp,
                                                    iL1, thr, rmx, iZ, Oh16);

  // 7) Output projection -> f32 d_out
  gemm_bias_kernel<2><<<gemmGrid, 128, 0, stream>>>(Oh16, Wo16, bo, out,
                                                    BATCH * SEQ, DMODEL, DMODEL);
}
